// KGramMLPSeqModel_30331059044622
// MI455X (gfx1250) — compile-verified, matrix operand order checked by reading, not run on hardware
//
#include <hip/hip_runtime.h>
#include <math.h>

#define SEQ    2048
#define BATCH  4
#define KCTX   3
#define EMBED  512
#define CIN    (KCTX * EMBED)   /* 1536 */
#define HID    256
#define VOCAB  32000
#define MROWS  (SEQ * BATCH)    /* 8192 */

typedef __attribute__((ext_vector_type(16))) _Float16 v16h;
typedef __attribute__((ext_vector_type(8)))  _Float16 v8h;
typedef __attribute__((ext_vector_type(8)))  float    v8f;

union V16U { v16h v; v8h h8[2]; };

// Load a 16-half WMMA A/B fragment slice for this lane from f16 memory.
// p points at element (row, kbase); elems 0-7 = p[0..7], elems 8-15 = p[16..23].
__device__ __forceinline__ v16h load_frag_h16(const _Float16* __restrict__ p) {
  V16U u;
  u.h8[0] = *(const v8h*)(p);
  u.h8[1] = *(const v8h*)(p + 16);
  return u.v;
}

// ---- one-time conversions into workspace ------------------------------------

// W1 (CIN x HID, f32) -> W1T (HID x CIN, f16), transposed for contiguous B frags
__global__ void cvt_w1(const float* __restrict__ W1, _Float16* __restrict__ W1T) {
  const int c = blockIdx.x;   // 0..CIN-1
  const int j = threadIdx.x;  // 0..HID-1
  W1T[(size_t)j * CIN + c] = (_Float16)W1[(size_t)c * HID + j];
}

// W2 (HID x VOCAB, f32) -> W2T (VOCAB x HID, f16)
__global__ void cvt_w2(const float* __restrict__ W2, _Float16* __restrict__ W2T) {
  const int n = blockIdx.x;   // 0..VOCAB-1
  const int k = threadIdx.x;  // 0..HID-1
  W2T[(size_t)n * HID + k] = (_Float16)W2[(size_t)k * VOCAB + n];
}

// emb (VOCAB x EMBED, f32) -> embH (VOCAB x EMBED, f16), same layout (row-major
// along k is what the A-fragment loader wants). Coalesced read + write.
__global__ void cvt_emb(const float* __restrict__ emb, _Float16* __restrict__ embH) {
  const size_t base = (size_t)blockIdx.x * EMBED;
  const int e = threadIdx.x;  // 0..255
  embH[base + e]       = (_Float16)emb[base + e];
  embH[base + e + 256] = (_Float16)emb[base + e + 256];
}

// ---- GEMM1: h = GELU(gather(embH, kgram) @ W1 + b1), h stored f16 (MROWS x HID)
// Grid: MROWS/64 blocks of 256 threads. Wave w handles N-tile [32w, 32w+32).
__global__ __launch_bounds__(256) void gemm1_gelu(
    const int* __restrict__ tokens, const _Float16* __restrict__ embH,
    const _Float16* __restrict__ W1T, const float* __restrict__ b1,
    _Float16* __restrict__ hbuf) {
  const int lane = threadIdx.x & 31;
  const int wave = threadIdx.x >> 5;       // 0..7
  const int l16  = lane & 15;
  const int hi8  = (lane >> 4) ? 8 : 0;
  const int m0   = blockIdx.x * 64;
  const int n0   = wave * 32;

  // token ids for this lane's 4 rows (one per 16-row M subtile), 3 context slots
  int tok[4][KCTX];
#pragma unroll
  for (int mt = 0; mt < 4; ++mt) {
    const int m = m0 + mt * 16 + l16;
    const int s = m >> 2, b = m & 3;
#pragma unroll
    for (int j = 0; j < KCTX; ++j) {
      const int sj = s - (KCTX - 1) + j;
      tok[mt][j] = (sj < 0) ? 0 : tokens[sj * BATCH + b];
    }
  }

  v8f acc[4][2];
#pragma unroll
  for (int nt = 0; nt < 2; ++nt) {
    const float bv = b1[n0 + nt * 16 + l16];
#pragma unroll
    for (int mt = 0; mt < 4; ++mt)
#pragma unroll
      for (int i = 0; i < 8; ++i) acc[mt][nt][i] = bv;
  }

#pragma unroll
  for (int j = 0; j < KCTX; ++j) {          // static j -> no dynamic reg indexing
    for (int ke = 0; ke < EMBED; ke += 32) {
      const int kk = j * EMBED + ke;
      v16h bfr[2];
#pragma unroll
      for (int nt = 0; nt < 2; ++nt)
        bfr[nt] = load_frag_h16(W1T + (size_t)(n0 + nt * 16 + l16) * CIN + kk + hi8);
#pragma unroll
      for (int mt = 0; mt < 4; ++mt) {
        const v16h a = load_frag_h16(embH + (size_t)tok[mt][j] * EMBED + ke + hi8);
#pragma unroll
        for (int nt = 0; nt < 2; ++nt)
          acc[mt][nt] = __builtin_amdgcn_wmma_f32_16x16x32_f16(
              false, a, false, bfr[nt], (short)0, acc[mt][nt], false, false);
      }
    }
  }

  // exact GELU then store f16
#pragma unroll
  for (int mt = 0; mt < 4; ++mt)
#pragma unroll
    for (int nt = 0; nt < 2; ++nt)
#pragma unroll
      for (int r = 0; r < 8; ++r) {
        const int m = m0 + mt * 16 + hi8 + r;
        const int n = n0 + nt * 16 + l16;
        const float x = acc[mt][nt][r];
        const float g = 0.5f * x * (1.0f + erff(x * 0.70710678118654752f));
        hbuf[(size_t)m * HID + n] = (_Float16)g;
      }
}

// ---- GEMM2: out = h @ W2 + b2   (MROWS x VOCAB, f32 out)
// Grid: (MROWS/128)*(VOCAB/256) blocks of 256 threads.
// Wave (wm,wn) = (wave>>2, wave&3) computes a 64x64 tile.
__global__ __launch_bounds__(256) void gemm2(
    const _Float16* __restrict__ hbuf, const _Float16* __restrict__ W2T,
    const float* __restrict__ b2, float* __restrict__ out) {
  const int lane = threadIdx.x & 31;
  const int wave = threadIdx.x >> 5;
  const int l16  = lane & 15;
  const int hi8  = (lane >> 4) ? 8 : 0;

  const int nblocks = VOCAB / 256;         // 125
  const int nb = blockIdx.x % nblocks;
  const int mb = blockIdx.x / nblocks;
  const int m0 = mb * 128 + (wave >> 2) * 64;
  const int n0 = nb * 256 + (wave & 3) * 64;

  v8f acc[4][4];
#pragma unroll
  for (int nt = 0; nt < 4; ++nt) {
    const float bv = b2[n0 + nt * 16 + l16];
#pragma unroll
    for (int mt = 0; mt < 4; ++mt)
#pragma unroll
      for (int i = 0; i < 8; ++i) acc[mt][nt][i] = bv;
  }

#pragma unroll
  for (int kk = 0; kk < HID; kk += 32) {   // 8 k-steps, fully unrolled
    v16h afr[4];
#pragma unroll
    for (int mt = 0; mt < 4; ++mt)
      afr[mt] = load_frag_h16(hbuf + (size_t)(m0 + mt * 16 + l16) * HID + kk + hi8);
#pragma unroll
    for (int nt = 0; nt < 4; ++nt) {
      const v16h b = load_frag_h16(W2T + (size_t)(n0 + nt * 16 + l16) * HID + kk + hi8);
#pragma unroll
      for (int mt = 0; mt < 4; ++mt)
        acc[mt][nt] = __builtin_amdgcn_wmma_f32_16x16x32_f16(
            false, afr[mt], false, b, (short)0, acc[mt][nt], false, false);
    }
  }

#pragma unroll
  for (int mt = 0; mt < 4; ++mt)
#pragma unroll
    for (int nt = 0; nt < 4; ++nt)
#pragma unroll
      for (int r = 0; r < 8; ++r) {
        const int m = m0 + mt * 16 + hi8 + r;
        const int n = n0 + nt * 16 + l16;
        out[(size_t)m * VOCAB + n] = acc[mt][nt][r];
      }
}

extern "C" void kernel_launch(void* const* d_in, const int* in_sizes, int n_in,
                              void* d_out, int out_size, void* d_ws, size_t ws_size,
                              hipStream_t stream) {
  const int*   tokens = (const int*)  d_in[0];
  const float* emb    = (const float*)d_in[1];
  const float* W1     = (const float*)d_in[2];
  const float* b1     = (const float*)d_in[3];
  const float* W2     = (const float*)d_in[4];
  const float* b2     = (const float*)d_in[5];
  float* out = (float*)d_out;

  // workspace layout (all 16B-aligned): W1T | W2T | embH | h
  char* ws = (char*)d_ws;
  const size_t w1t_bytes = (size_t)HID * CIN * sizeof(_Float16);      // 768 KB
  const size_t w2t_bytes = (size_t)VOCAB * HID * sizeof(_Float16);    // 16 MB
  const size_t emb_bytes = (size_t)VOCAB * EMBED * sizeof(_Float16);  // 32.8 MB
  _Float16* W1T  = (_Float16*)(ws);
  _Float16* W2T  = (_Float16*)(ws + w1t_bytes);
  _Float16* embH = (_Float16*)(ws + w1t_bytes + w2t_bytes);
  _Float16* hbuf = (_Float16*)(ws + w1t_bytes + w2t_bytes + emb_bytes); // 4 MB

  cvt_w1 <<<CIN,   HID, 0, stream>>>(W1, W1T);
  cvt_w2 <<<VOCAB, HID, 0, stream>>>(W2, W2T);
  cvt_emb<<<VOCAB, 256, 0, stream>>>(emb, embH);
  gemm1_gelu<<<MROWS / 64, 256, 0, stream>>>(tokens, embH, W1T, b1, hbuf);
  gemm2<<<(MROWS / 128) * (VOCAB / 256), 256, 0, stream>>>(hbuf, W2T, b2, out);
}
